// Block_13752485281967
// MI455X (gfx1250) — compile-verified, matrix operand order checked by reading, not run on hardware
//
#include <hip/hip_runtime.h>
#include <hip/hip_bf16.h>

// ---------------------------------------------------------------------------
// MI455X (gfx1250):
//   q,k,v = tokens @ W{q,k,v} + b        [8192,4096]
//   att   = softmax(q^T @ k)             [4096,4096]
//   out   = att @ v^T                    [4096,8192]
//   final = out @ Wp + bp                [4096,8192]  (fp32 output)
// ~1.9 TFLOP vs ~0.7GB traffic -> compute bound -> bf16 WMMA, f32 accumulate.
// Double-buffered LDS; bf16 operands staged with GLOBAL_LOAD_ASYNC_TO_LDS_B128
// (ASYNCcnt) so tile k+1 copies overlap the WMMA stream on tile k.
// 64x64 wave tile: 16 v_wmma per K-step from 8 fragment loads (4x operand
// reuse) -> 1.0 ds_load_b128 per WMMA.
// ---------------------------------------------------------------------------

#define T_DIM 8192
#define E_DIM 4096

typedef __attribute__((ext_vector_type(16))) __bf16 v16bf;
typedef __attribute__((ext_vector_type(8)))  float  v8f;

union Frag16 {
  v16bf v;
  unsigned int u[8];
};

__device__ __forceinline__ unsigned short f2bf_bits(float f) {
  unsigned int u = __builtin_bit_cast(unsigned int, f);
  u += 0x7fffu + ((u >> 16) & 1u);          // round-to-nearest-even
  return (unsigned short)(u >> 16);
}
__device__ __forceinline__ unsigned int pack_bf2(float lo, float hi) {
  return (unsigned int)f2bf_bits(lo) | ((unsigned int)f2bf_bits(hi) << 16);
}

__device__ __forceinline__ void store_c(float* p, float v) { *p = v; }
__device__ __forceinline__ void store_c(unsigned short* p, float v) { *p = f2bf_bits(v); }

// 16B global -> 16B LDS, no VGPR data path, tracked by ASYNCcnt.
// LDS operand = low 32 bits of the generic shared pointer (aperture layout:
// addr[31:0] is the LDS byte offset).
__device__ __forceinline__ void async_copy_b128(void* lds, const void* g) {
  asm volatile("global_load_async_to_lds_b128 %0, %1, off"
               :
               : "v"((unsigned int)(unsigned long long)lds),
                 "v"((unsigned long long)g)
               : "memory");
}

// ---------------------------------------------------------------------------
// Tiled bf16 WMMA GEMM:  C[M,N] = op(A)[M,K] * op(B)[K,N] (+ bias[n])
//   TRANS_A: A(m,k) = A[k*lda + m]   (q^T @ k)
//   TRANS_B: B(k,n) = B[n*ldb + k]   (att @ v^T; contiguous in v rows)
// Block tile 128x256, BK=32, 256 threads = 8 wave32s (2x4 wave grid),
// each wave owns a 64x64 patch = 4x4 v_wmma_f32_16x16x32_bf16 accumulators.
// M % 128 == 0, N % 256 == 0, K % 32 == 0 (true for all six GEMMs here).
// ---------------------------------------------------------------------------
template <int TRANS_A, int TRANS_B, int HAS_BIAS,
          typename AT, typename BT, typename CT>
__global__ __launch_bounds__(256) void wmma_gemm(
    const AT* __restrict__ A, int lda,
    const BT* __restrict__ B, int ldb,
    const float* __restrict__ bias,
    CT* __restrict__ C, int ldc,
    int M, int N, int K) {
  constexpr bool A_BF16  = (sizeof(AT) == 2);
  constexpr bool B_BF16  = (sizeof(BT) == 2);
  constexpr bool ASYNC_A = A_BF16 && !TRANS_A;  // contiguous bf16 rows
  constexpr bool ASYNC_B = B_BF16 && TRANS_B;   // v rows == N-major LDS layout

  // +8 pad: 80B row stride (16B aligned, staggered across 64 LDS banks).
  __shared__ __align__(16) unsigned short Asb[2][128][40];  // As[m][k]
  __shared__ __align__(16) unsigned short Bsb[2][256][40];  // Bs[n][k] N-major

  const int tid  = threadIdx.x;
  const int lane = tid & 31;
  const int wave = tid >> 5;
  const int wm   = (wave & 1) << 6;  // 0,64
  const int wn   = (wave >> 1) << 6; // 0,64,128,192
  const int m0   = blockIdx.y * 128;
  const int n0   = blockIdx.x * 256;

  const int half = lane >> 4;        // ISA 16-bit layout lane-half K split
  const int kb0  = half << 3;        // 0 or 8

  v8f acc[4][4];
#pragma unroll
  for (int mi = 0; mi < 4; ++mi)
#pragma unroll
    for (int ni = 0; ni < 4; ++ni)
      acc[mi][ni] = (v8f){};

  // ---- stage one 128x32 A-tile + 32x256 B-tile into buffer `buf` ---------
  auto stage = [&](int buf, int k0) {
    unsigned short(*As)[40] = Asb[buf];
    unsigned short(*Bs)[40] = Bsb[buf];
    // A operand (128 rows)
    if constexpr (ASYNC_A) {
      for (int i = tid; i < 128 * 4; i += 256) {           // 16B per item
        const int r = i >> 2, c8 = (i & 3) << 3;
        async_copy_b128(&As[r][c8], &A[(long)(m0 + r) * lda + (k0 + c8)]);
      }
    } else if constexpr (A_BF16) {                         // bf16, TRANS_A
      for (int i = tid; i < 512; i += 256) {               // 8 m per item
        const int mg = (i >> 5) << 3, kk = i & 31;
        const uint4 u = *(const uint4*)&A[(long)(k0 + kk) * lda + (m0 + mg)];
        const unsigned short* s = (const unsigned short*)&u;
#pragma unroll
        for (int t = 0; t < 8; ++t) As[mg + t][kk] = s[t];
      }
    } else if constexpr (!TRANS_A) {                       // fp32, no trans
      for (int i = tid; i < 1024; i += 256) {              // 4 k per item
        const int r = i >> 3, c4 = (i & 7) << 2;
        const float4 f = *(const float4*)&A[(long)(m0 + r) * lda + (k0 + c4)];
        *(unsigned int*)&As[r][c4]     = pack_bf2(f.x, f.y);
        *(unsigned int*)&As[r][c4 + 2] = pack_bf2(f.z, f.w);
      }
    } else {                                               // fp32 trans (unused)
      for (int i = tid; i < 4096; i += 256) {
        const int r = i >> 5, c = i & 31;
        As[r][c] = f2bf_bits((float)A[(long)(k0 + c) * lda + (m0 + r)]);
      }
    }
    // B operand (256 rows, LDS is N-major: Bs[n][k])
    if constexpr (ASYNC_B) {
      for (int i = tid; i < 256 * 4; i += 256) {           // 16B per item
        const int r = i >> 2, c8 = (i & 3) << 3;
        async_copy_b128(&Bs[r][c8], &B[(long)(n0 + r) * ldb + (k0 + c8)]);
      }
    } else if constexpr (B_BF16) {                         // bf16, no trans
      for (int i = tid; i < 1024; i += 256) {              // 8 n per item
        const int ng = (i >> 5) << 3, kk = i & 31;
        const uint4 u = *(const uint4*)&B[(long)(k0 + kk) * ldb + (n0 + ng)];
        const unsigned short* s = (const unsigned short*)&u;
#pragma unroll
        for (int t = 0; t < 8; ++t) Bs[ng + t][kk] = s[t];
      }
    } else if constexpr (!TRANS_B) {                       // fp32, no trans
      for (int i = tid; i < 2048; i += 256) {              // 4 n per item
        const int ng = (i >> 5) << 2, kk = i & 31;
        const float4 f = *(const float4*)&B[(long)(k0 + kk) * ldb + (n0 + ng)];
        Bs[ng + 0][kk] = f2bf_bits(f.x);
        Bs[ng + 1][kk] = f2bf_bits(f.y);
        Bs[ng + 2][kk] = f2bf_bits(f.z);
        Bs[ng + 3][kk] = f2bf_bits(f.w);
      }
    } else {                                               // fp32 trans (unused)
      for (int i = tid; i < 8192; i += 256) {
        const int r = i >> 5, c = i & 31;
        Bs[r][c] = f2bf_bits((float)B[(long)(n0 + r) * ldb + (k0 + c)]);
      }
    }
  };

  auto tile_sync = [&]() {
    if constexpr (ASYNC_A || ASYNC_B)
      asm volatile("s_wait_asynccnt 0x0" ::: "memory");    // our async copies done
    __syncthreads();
  };

  stage(0, 0);
  tile_sync();

  int cur = 0;
  for (int k0 = 0; k0 < K; k0 += 32) {
    if (k0 + 32 < K) stage(cur ^ 1, k0 + 32);  // overlap copy with WMMA below

    unsigned short(*As)[40] = Asb[cur];
    unsigned short(*Bs)[40] = Bsb[cur];

    // fragment loads per ISA 16-bit 16x32 layout:
    // lane L: row/col = L&15 ; VGPR j holds K = (j>=4?16:0) + (L>>4)*8 + 2*(j&3)
    Frag16 a[4], b[4];
#pragma unroll
    for (int mi = 0; mi < 4; ++mi) {
      const unsigned short* p = &As[wm + (mi << 4) + (lane & 15)][0];
#pragma unroll
      for (int j = 0; j < 8; ++j) {
        const int kk = ((j & 4) << 2) + kb0 + ((j & 3) << 1);
        a[mi].u[j] = *(const unsigned int*)(p + kk);
      }
    }
#pragma unroll
    for (int ni = 0; ni < 4; ++ni) {
      const unsigned short* p = &Bs[wn + (ni << 4) + (lane & 15)][0];
#pragma unroll
      for (int j = 0; j < 8; ++j) {
        const int kk = ((j & 4) << 2) + kb0 + ((j & 3) << 1);
        b[ni].u[j] = *(const unsigned int*)(p + kk);
      }
    }

#pragma unroll
    for (int mi = 0; mi < 4; ++mi)
#pragma unroll
      for (int ni = 0; ni < 4; ++ni)
        acc[mi][ni] = __builtin_amdgcn_wmma_f32_16x16x32_bf16(
            false, a[mi].v, false, b[ni].v, (short)0, acc[mi][ni],
            false, false);

    tile_sync();  // next buffer staged; current buffer free for reuse
    cur ^= 1;
  }

  // ---- epilogue: D layout lane L -> n = L&15, m = j + 8*(L>>4) -----------
#pragma unroll
  for (int mi = 0; mi < 4; ++mi) {
#pragma unroll
    for (int ni = 0; ni < 4; ++ni) {
      const int col  = n0 + wn + (ni << 4) + (lane & 15);
      const float bv = HAS_BIAS ? bias[col] : 0.0f;
#pragma unroll
      for (int j = 0; j < 8; ++j) {
        const int row = m0 + wm + (mi << 4) + (half << 3) + j;
        store_c(&C[(long)row * ldc + col], acc[mi][ni][j] + bv);
      }
    }
  }
}

// ---------------------------------------------------------------------------
// Row softmax over S[E,E] (f32), writes att as bf16.
// ---------------------------------------------------------------------------
__global__ __launch_bounds__(256) void softmax_row_kernel(
    const float* __restrict__ S, unsigned short* __restrict__ att, int N) {
  __shared__ float red[256];
  const int tid = threadIdx.x;
  const float* s = S + (long)blockIdx.x * N;

  float mx = -3.402823466e38f;
  for (int i = tid; i < N; i += 256) mx = fmaxf(mx, s[i]);
  red[tid] = mx;
  __syncthreads();
  for (int off = 128; off > 0; off >>= 1) {
    if (tid < off) red[tid] = fmaxf(red[tid], red[tid + off]);
    __syncthreads();
  }
  const float m = red[0];
  __syncthreads();

  float sum = 0.0f;
  for (int i = tid; i < N; i += 256) sum += __expf(s[i] - m);
  red[tid] = sum;
  __syncthreads();
  for (int off = 128; off > 0; off >>= 1) {
    if (tid < off) red[tid] += red[tid + off];
    __syncthreads();
  }
  const float inv = 1.0f / red[0];

  unsigned short* o = att + (long)blockIdx.x * N;
  for (int i = tid; i < N; i += 256) o[i] = f2bf_bits(__expf(s[i] - m) * inv);
}

// ---------------------------------------------------------------------------
extern "C" void kernel_launch(void* const* d_in, const int* in_sizes, int n_in,
                              void* d_out, int out_size, void* d_ws,
                              size_t ws_size, hipStream_t stream) {
  (void)in_sizes; (void)n_in; (void)out_size; (void)ws_size;
  const float* tokens = (const float*)d_in[0];
  const float* Wq = (const float*)d_in[1];
  const float* bq = (const float*)d_in[2];
  const float* Wk = (const float*)d_in[3];
  const float* bk = (const float*)d_in[4];
  const float* Wv = (const float*)d_in[5];
  const float* bv = (const float*)d_in[6];
  const float* Wp = (const float*)d_in[7];
  const float* bp = (const float*)d_in[8];
  float* out = (float*)d_out;

  const int T = T_DIM, E = E_DIM;

  // Workspace layout (~352 MB): q,k,v,out2 bf16; scores f32; att bf16.
  unsigned short* qb = (unsigned short*)d_ws;
  unsigned short* kb = qb + (size_t)T * E;
  unsigned short* vb = kb + (size_t)T * E;
  float* S             = (float*)(vb + (size_t)T * E);
  unsigned short* attb = (unsigned short*)(S + (size_t)E * E);
  unsigned short* outb = attb + (size_t)E * E;

  dim3 blk(256);

  // 1) q/k/v = tokens @ W + b : [T,E] (bf16)
  dim3 gq(E / 256, T / 128);
  wmma_gemm<0, 0, 1, float, float, unsigned short>
      <<<gq, blk, 0, stream>>>(tokens, E, Wq, E, bq, qb, E, T, E, E);
  wmma_gemm<0, 0, 1, float, float, unsigned short>
      <<<gq, blk, 0, stream>>>(tokens, E, Wk, E, bk, kb, E, T, E, E);
  wmma_gemm<0, 0, 1, float, float, unsigned short>
      <<<gq, blk, 0, stream>>>(tokens, E, Wv, E, bv, vb, E, T, E, E);

  // 2) S = q^T @ k : [E,E] (f32), K = T
  dim3 gs(E / 256, E / 128);
  wmma_gemm<1, 0, 0, unsigned short, unsigned short, float>
      <<<gs, blk, 0, stream>>>(qb, E, kb, E, nullptr, S, E, E, E, T);

  // 3) att = softmax(S) rows : bf16
  softmax_row_kernel<<<E, blk, 0, stream>>>(S, attb, E);

  // 4) out2 = att @ v^T : [E,T] (bf16), K = E   (async A and async B paths)
  dim3 go(T / 256, E / 128);
  wmma_gemm<0, 1, 0, unsigned short, unsigned short, unsigned short>
      <<<go, blk, 0, stream>>>(attb, E, vb, E, nullptr, outb, T, E, T, E);

  // 5) final = out2 @ Wp + bp : [E,T] (f32 -> d_out), K = T
  wmma_gemm<0, 0, 1, unsigned short, float, float>
      <<<go, blk, 0, stream>>>(outb, T, Wp, T, bp, out, T, E, T, T);
}
